// SessionGraph_35940286333643
// MI455X (gfx1250) — matrix operand order repaired and checked by reading, compiled, not run on hardware
//
#include <hip/hip_runtime.h>
#include <hip/hip_bf16.h>

// ---------------------------------------------------------------------------
// Problem constants (from reference)
// ---------------------------------------------------------------------------
#define BATCH 1024
#define L     20
#define HID   128
#define M_TOK (BATCH * L)          // 20480 rows
#define N1    640                  // [h1 | h2 | gh] columns
#define N2    384                  // gi columns
#define K1    128
#define K2    256

typedef __attribute__((ext_vector_type(16))) _Float16 v16h;
typedef __attribute__((ext_vector_type(8)))  float    v8f;
typedef __attribute__((ext_vector_type(4)))  _Float16 v4h;

// ---------------------------------------------------------------------------
// Kernel 0: pre-swizzle WMMA B operands + fused bias
//   Bh  : K=128 x N=640  =  [W_in^T | W_out^T | w_hh^T]   (f16, WMMA-B layout)
//   Bi  : K=256 x N=384  =  w_ih^T                        (f16, WMMA-B layout)
//   bias1[640] = [b_in | b_out | b_hh]
// B per-lane layout used here (contiguous k, same as A): column n = lane%16,
//   lanes 0-15 hold K = kc*32+0..15, lanes 16-31 hold K = kc*32+16..31,
//   stored as 16 consecutive halves per lane -> one 32B contiguous chunk.
// ---------------------------------------------------------------------------
__global__ void prep_kernel(const float* __restrict__ W_in,
                            const float* __restrict__ W_out,
                            const float* __restrict__ w_hh,
                            const float* __restrict__ w_ih,
                            const float* __restrict__ b_in,
                            const float* __restrict__ b_out,
                            const float* __restrict__ b_hh,
                            _Float16* __restrict__ Bh,
                            _Float16* __restrict__ Bi,
                            float* __restrict__ bias1)
{
    const int NBH = (N1 / 16) * (K1 / 32) * 32 * 16;   // 81920
    const int NBI = (N2 / 16) * (K2 / 32) * 32 * 16;   // 98304
    int tid = blockIdx.x * blockDim.x + threadIdx.x;
    if (tid < NBH) {
        int e    = tid & 15;
        int lane = (tid >> 4) & 31;
        int kt   = tid >> 9;            // tn*(K1/32) + kc
        int kc   = kt & 3;
        int tn   = kt >> 2;
        int n    = tn * 16 + (lane & 15);
        int k    = kc * 32 + ((lane & 16) ? 16 : 0) + e;
        float v;
        if (n < 128)      v = W_in[n * 128 + k];
        else if (n < 256) v = W_out[(n - 128) * 128 + k];
        else              v = w_hh[(n - 256) * 128 + k];
        Bh[tid] = (_Float16)v;
    } else if (tid < NBH + NBI) {
        int t2   = tid - NBH;
        int e    = t2 & 15;
        int lane = (t2 >> 4) & 31;
        int kt   = t2 >> 9;             // tn*(K2/32) + kc
        int kc   = kt & 7;
        int tn   = kt >> 3;
        int n    = tn * 16 + (lane & 15);
        int k    = kc * 32 + ((lane & 16) ? 16 : 0) + e;
        Bi[t2] = (_Float16)w_ih[n * 256 + k];
    } else if (tid < NBH + NBI + N1) {
        int n = tid - NBH - NBI;
        bias1[n] = (n < 128) ? b_in[n] : (n < 256) ? b_out[n - 128] : b_hh[n - 256];
    }
}

// ---------------------------------------------------------------------------
// Kernel 1: gather + per-row L2 normalize (only the 20480 live rows of emb!),
//           time_tab gather -> d_out, hidden_avg -> d_out.
// One block (256 thr = 8 waves) per batch item; one wave per session row.
// ---------------------------------------------------------------------------
__global__ void gather_norm_kernel(const int*   __restrict__ inputs,
                                   const int*   __restrict__ times,
                                   const float* __restrict__ emb,
                                   const float* __restrict__ time_tab,
                                   float*    __restrict__ Hf32,
                                   _Float16* __restrict__ Hf16,
                                   float*    __restrict__ out_time,
                                   float*    __restrict__ out_avg)
{
    __shared__ float s_avg[HID];
    const int b    = blockIdx.x;
    const int tid  = threadIdx.x;
    const int wave = tid >> 5;
    const int lane = tid & 31;
    if (tid < HID) s_avg[tid] = 0.f;
    __syncthreads();

    float4 acc = {0.f, 0.f, 0.f, 0.f};
    for (int l = wave; l < L; l += 8) {
        const int row = b * L + l;
        // --- embedding gather + normalize ---
        const int idx = inputs[row];
        float4 v = ((const float4*)(emb + (size_t)idx * HID))[lane];
        float ss = v.x * v.x + v.y * v.y + v.z * v.z + v.w * v.w;
        #pragma unroll
        for (int off = 16; off > 0; off >>= 1) ss += __shfl_xor(ss, off, 32);
        const float s = rsqrtf(ss);
        v.x *= s; v.y *= s; v.z *= s; v.w *= s;
        ((float4*)(Hf32 + (size_t)row * HID))[lane] = v;
        v4h hv; hv[0] = (_Float16)v.x; hv[1] = (_Float16)v.y;
                hv[2] = (_Float16)v.z; hv[3] = (_Float16)v.w;
        ((v4h*)(Hf16 + (size_t)row * HID))[lane] = hv;
        acc.x += v.x; acc.y += v.y; acc.z += v.z; acc.w += v.w;
        // --- time embedding gather ---
        const int t = times[row];
        float4 tv = ((const float4*)(time_tab + (size_t)t * HID))[lane];
        ((float4*)(out_time + (size_t)row * HID))[lane] = tv;
    }
    atomicAdd(&s_avg[lane * 4 + 0], acc.x);
    atomicAdd(&s_avg[lane * 4 + 1], acc.y);
    atomicAdd(&s_avg[lane * 4 + 2], acc.z);
    atomicAdd(&s_avg[lane * 4 + 3], acc.w);
    __syncthreads();
    if (tid < HID) out_avg[b * HID + tid] = s_avg[tid] * (1.0f / (float)L);
}

// ---------------------------------------------------------------------------
// WMMA GEMM: C[M x N] = A[M x K](f16,row-major) * Bswz(f16,pre-swizzled) + bias
//
// Block = 8 waves covering 8 M-tiles x ONE 64-column strip:
//   - the strip's pre-swizzled B (64 x K halves, contiguous in Bswz) is
//     cooperatively staged in LDS once and reused by all 8 waves,
//   - each wave preloads ALL its A fragments into registers before the
//     barrier, so the fully-unrolled compute loop runs WMMAs back-to-back
//     against ds_load_b128, with no global-load waits inside.
//
// A fragment per-lane (16x16x32 f16): row m = lane%16,
//   lanes 0-15: K = kc*32 + 0..15, lanes 16-31: K = kc*32 + 16..31 (contiguous, as in B)
// C/D layout: VGPR j -> row j + 8*(lane>=16), col n = lane%16.
// ---------------------------------------------------------------------------
template<int K_CHUNKS, int N>
__global__ __launch_bounds__(256)
void gemm_wmma_kernel(const _Float16* __restrict__ A,
                      const _Float16* __restrict__ Bswz,
                      const float*    __restrict__ bias,
                      float*          __restrict__ C)
{
    constexpr int K            = K_CHUNKS * 32;
    constexpr int N_STRIPS     = N / 64;
    constexpr int STRIP_HALVES = 4 * K_CHUNKS * 512;   // 64 cols x K halves
    __shared__ _Float16 sB[STRIP_HALVES];

    const int tid     = threadIdx.x;
    const int wave    = tid >> 5;
    const int lane    = tid & 31;
    const int strip_n = blockIdx.x % N_STRIPS;
    const int m_blk   = blockIdx.x / N_STRIPS;
    const int m0      = (m_blk * 8 + wave) * 16;
    const int n0      = strip_n * 64;

    // --- cooperative LDS fill: strip of Bswz is contiguous & frag-swizzled ---
    {
        const uint4* src = (const uint4*)(Bswz + (size_t)strip_n * STRIP_HALVES);
        uint4*       dst = (uint4*)sB;
        constexpr int NV = STRIP_HALVES / 8;           // uint4 elements
        #pragma unroll
        for (int i = 0; i < NV / 256; ++i)
            dst[tid + i * 256] = src[tid + i * 256];
    }

    // --- preload ALL A fragments for this wave (independent of the barrier) ---
    union Frag { uint4 u[2]; v16h h; };
    Frag af[K_CHUNKS];
    const _Float16* Abase =
        A + (size_t)(m0 + (lane & 15)) * K + ((lane & 16) ? 16 : 0);
    #pragma unroll
    for (int kc = 0; kc < K_CHUNKS; ++kc) {
        af[kc].u[0] = *(const uint4*)(Abase + kc * 32);
        af[kc].u[1] = *(const uint4*)(Abase + kc * 32 + 8);
    }

    __syncthreads();

    v8f c[4];
    const v8f z = {0.f, 0.f, 0.f, 0.f, 0.f, 0.f, 0.f, 0.f};
    #pragma unroll
    for (int t = 0; t < 4; ++t) c[t] = z;

    #pragma unroll
    for (int kc = 0; kc < K_CHUNKS; ++kc) {
        #pragma unroll
        for (int t = 0; t < 4; ++t) {
            Frag bf;
            const _Float16* bp = sB + ((t * K_CHUNKS + kc) * 32 + lane) * 16;
            bf.u[0] = *(const uint4*)(bp);
            bf.u[1] = *(const uint4*)(bp + 8);
            c[t] = __builtin_amdgcn_wmma_f32_16x16x32_f16(
                false, af[kc].h, false, bf.h, (short)0, c[t], false, false);
        }
    }

    const int ncol  = lane & 15;
    const int rbase = (lane >> 4) * 8;
    #pragma unroll
    for (int t = 0; t < 4; ++t) {
        const int n  = n0 + t * 16 + ncol;
        const float bv = bias[n];
        #pragma unroll
        for (int j = 0; j < 8; ++j) {
            const int m = m0 + rbase + j;
            C[(size_t)m * N + n] = c[t][j] + bv;
        }
    }
}

// ---------------------------------------------------------------------------
// Kernel 3: batched 20x20 adjacency einsum (fp32 in LDS; A kept full precision)
//   input_in  = A[:, :, :L]   @ h1 + b_iah
//   input_out = A[:, :, L:2L] @ h2 + b_oah
//   -> inputs_cat f16 [M_TOK x 256] (A operand of GEMM2)
// ---------------------------------------------------------------------------
__global__ void einsum_kernel(const float* __restrict__ Aadj,   // [B, L, 2L]
                              const float* __restrict__ C1,     // [M_TOK, 640]
                              const float* __restrict__ b_iah,
                              const float* __restrict__ b_oah,
                              _Float16*    __restrict__ cat)    // [M_TOK, 256]
{
    __shared__ float h1s[L * HID];
    __shared__ float h2s[L * HID];
    __shared__ float As[L * 2 * L];
    const int b = blockIdx.x, tid = threadIdx.x;
    for (int idx = tid; idx < L * HID; idx += 256) {
        const int j = idx >> 7, h = idx & 127;
        const float* row = C1 + (size_t)(b * L + j) * N1;
        h1s[idx] = row[h];
        h2s[idx] = row[HID + h];
    }
    for (int idx = tid; idx < L * 2 * L; idx += 256)
        As[idx] = Aadj[(size_t)b * (L * 2 * L) + idx];
    __syncthreads();
    for (int idx = tid; idx < L * HID; idx += 256) {
        const int i = idx >> 7, h = idx & 127;
        float accin  = b_iah[h];
        float accout = b_oah[h];
        const float* arow = As + i * (2 * L);
        #pragma unroll
        for (int j = 0; j < L; ++j) {
            accin  += arow[j]     * h1s[j * HID + h];
            accout += arow[L + j] * h2s[j * HID + h];
        }
        _Float16* dst = cat + (size_t)(b * L + i) * 256;
        dst[h]       = (_Float16)accin;
        dst[HID + h] = (_Float16)accout;
    }
}

// ---------------------------------------------------------------------------
// Kernel 5: fused GRU gates -> final hidden (d_out section 0)
// ---------------------------------------------------------------------------
__global__ void gate_kernel(const float* __restrict__ C1,    // gh at cols 256..639
                            const float* __restrict__ C2,    // gi [M_TOK x 384]
                            const float* __restrict__ Hf32,  // normalized hidden
                            float*       __restrict__ out)
{
    const int tid = blockIdx.x * blockDim.x + threadIdx.x;
    if (tid >= M_TOK * HID) return;
    const int m = tid >> 7, h = tid & 127;
    const float* gi = C2 + (size_t)m * N2;
    const float* gh = C1 + (size_t)m * N1 + 256;
    float rg = gi[h]         + gh[h];
    float ig = gi[HID + h]   + gh[HID + h];
    float gin = gi[2 * HID + h];
    float ghn = gh[2 * HID + h];
    rg = 1.f / (1.f + expf(-rg));
    ig = 1.f / (1.f + expf(-ig));
    const float ng = tanhf(gin + rg * ghn);
    const float hd = Hf32[tid];
    out[tid] = ng + ig * (hd - ng);
}

// ---------------------------------------------------------------------------
extern "C" void kernel_launch(void* const* d_in, const int* in_sizes, int n_in,
                              void* d_out, int out_size, void* d_ws, size_t ws_size,
                              hipStream_t stream) {
    const int*   inputs   = (const int*)  d_in[0];
    const float* Aadj     = (const float*)d_in[1];
    const int*   times    = (const int*)  d_in[2];
    const float* emb      = (const float*)d_in[3];
    const float* time_tab = (const float*)d_in[4];
    const float* W_in     = (const float*)d_in[5];
    const float* b_in     = (const float*)d_in[6];
    const float* W_out    = (const float*)d_in[7];
    const float* b_out    = (const float*)d_in[8];
    const float* w_ih     = (const float*)d_in[9];
    const float* b_ih     = (const float*)d_in[10];
    const float* w_hh     = (const float*)d_in[11];
    const float* b_hh     = (const float*)d_in[12];
    const float* b_iah    = (const float*)d_in[13];
    const float* b_oah    = (const float*)d_in[14];

    // workspace carve-up
    char* ws = (char*)d_ws;
    size_t off = 0;
    auto carve = [&](size_t bytes) -> void* {
        void* p = ws + off;
        off += (bytes + 255) & ~(size_t)255;
        return p;
    };
    float*    Hf32  = (float*)   carve((size_t)M_TOK * HID * 4);
    _Float16* Hf16  = (_Float16*)carve((size_t)M_TOK * HID * 2);
    _Float16* Bh    = (_Float16*)carve((size_t)K1 * N1 * 2);
    _Float16* Bi    = (_Float16*)carve((size_t)K2 * N2 * 2);
    float*    bias1 = (float*)   carve((size_t)N1 * 4);
    float*    C1    = (float*)   carve((size_t)M_TOK * N1 * 4);
    _Float16* catF  = (_Float16*)carve((size_t)M_TOK * K2 * 2);
    float*    C2    = (float*)   carve((size_t)M_TOK * N2 * 4);

    // output sections (flat, in reference return order)
    float* out_hidden = (float*)d_out;
    float* out_time   = out_hidden + (size_t)M_TOK * HID;
    float* out_avg    = out_time   + (size_t)M_TOK * HID;

    // 0: weight pre-swizzle (tiny)
    {
        const int total = (N1 / 16) * (K1 / 32) * 512 + (N2 / 16) * (K2 / 32) * 512 + N1;
        prep_kernel<<<(total + 255) / 256, 256, 0, stream>>>(
            W_in, W_out, w_hh, w_ih, b_in, b_out, b_hh, Bh, Bi, bias1);
    }
    // 1: gather + normalize (only live rows) + time gather + avg
    gather_norm_kernel<<<BATCH, 256, 0, stream>>>(
        inputs, times, emb, time_tab, Hf32, Hf16, out_time, out_avg);
    // 2: C1 = Hn @ [W_in^T|W_out^T|w_hh^T] + [b_in|b_out|b_hh]
    //    blocks: (20480/16/8) * (640/64) = 160 * 10 = 1600
    gemm_wmma_kernel<K1 / 32, N1><<<(M_TOK / 128) * (N1 / 64), 256, 0, stream>>>(
        Hf16, Bh, bias1, C1);
    // 3: adjacency einsum -> inputs_cat (f16)
    einsum_kernel<<<BATCH, 256, 0, stream>>>(Aadj, C1, b_iah, b_oah, catF);
    // 4: C2 = cat @ w_ih^T + b_ih   (160 * 6 = 960 blocks)
    gemm_wmma_kernel<K2 / 32, N2><<<(M_TOK / 128) * (N2 / 64), 256, 0, stream>>>(
        catF, Bi, b_ih, C2);
    // 5: fused GRU gates -> final hidden
    gate_kernel<<<(M_TOK * HID + 255) / 256, 256, 0, stream>>>(C1, C2, Hf32, out_hidden);
}
